// WaveConv_38053410242673
// MI455X (gfx1250) — compile-verified, matrix operand order checked by reading, not run on hardware
//
#include <hip/hip_runtime.h>
#include <hip/hip_bf16.h>
#include <stdint.h>

// ---------------------------------------------------------------------------
// WaveConv (spectral graph wavelet conv), refactored:
//   out = U * ( [h.xt | g1.xt .. g5.xt] @ Wcat' ) + const,  xt = U^T x
// bf16 WMMA (v_wmma_f32_16x16x32_bf16) for all three GEMM stages.
// Round 3: fp32->bf16 via native fptrunc casts (compiler selects v_cvt*bf16)
// instead of hand-rolled RNE bit math that bloated VALU.
// ---------------------------------------------------------------------------

#define Bsz   2
#define Nn    4096
#define Dd    128
#define RHO   5
#define Jj    5
#define Cc    (6 * Dd)      // 768 concatenated filter channels
#define AS_STRIDE 40        // LDS row stride in ushorts (80B = 5*16B, 16B aligned)

typedef unsigned short u16;
typedef __bf16 bf16_t;
typedef bf16_t v16bf __attribute__((ext_vector_type(16)));
typedef bf16_t bf16x2 __attribute__((ext_vector_type(2)));
typedef float  f32x8 __attribute__((ext_vector_type(8)));
typedef u16    u16x8 __attribute__((ext_vector_type(8)));

union FragU { v16bf v; u16x8 h[2]; };

// fp32 -> bf16 via native fptrunc (backend picks v_cvt*bf16_f32)
__device__ __forceinline__ u16 f2bf(float f) {
    return __builtin_bit_cast(u16, (bf16_t)f);
}

// pack two fp32 -> 2x bf16 in one dword
__device__ __forceinline__ uint32_t pack_bf2(float lo, float hi) {
#if __has_builtin(__builtin_amdgcn_cvt_pk_bf16_f32)
    bf16x2 v = __builtin_amdgcn_cvt_pk_bf16_f32(lo, hi);
    return __builtin_bit_cast(uint32_t, v);
#else
    bf16x2 v = { (bf16_t)lo, (bf16_t)hi };
    return __builtin_bit_cast(uint32_t, v);
#endif
}

__device__ __forceinline__ f32x8 wmma_bf16(v16bf a, v16bf b, f32x8 c) {
    // 8 args: (neg_a, A, neg_b, B, c_mod, C, reuse_a, reuse_b)
    return __builtin_amdgcn_wmma_f32_16x16x32_bf16(
        false, a, false, b, (short)0, c, false, false);
}

// Chebyshev combos exactly mirroring the reference recurrences.
__device__ __forceinline__ float cheb_h(float e, const float* bc) {
    float To = 1.0f, Te = e;
    float s = bc[0] * To;
#pragma unroll
    for (int i = 1; i < RHO; ++i) {
        To = 2.0f * e * Te - To;
        Te = 2.0f * e * To - Te;
        s += bc[i] * To;
    }
    return s;
}
__device__ __forceinline__ float cheb_g(float sx, const float* ac) {
    float To = 1.0f, Te = sx;
    float s = ac[0] * Te;
#pragma unroll
    for (int i = 1; i < RHO; ++i) {
        To = 2.0f * sx * Te - To;
        Te = 2.0f * sx * To - Te;
        s += ac[i] * Te;
    }
    return s;
}

// ---------------------------------------------------------------------------
// 1) ebar[b][d] = mean_n eigvs[b][n][d]
__global__ void k_colmean(const float* __restrict__ eigvs, float* __restrict__ ebar) {
    const int b = blockIdx.x, d = threadIdx.x;
    const float* p = eigvs + (size_t)b * Nn * Dd + d;
    float s = 0.0f;
    for (int n = 0; n < Nn; ++n) s += p[(size_t)n * Dd];
    ebar[b * Dd + d] = s * (1.0f / (float)Nn);
}

// ---------------------------------------------------------------------------
// 2) x -> bf16, transposed per batch: xbf_t[(b*128+e)*4096 + m]
__global__ void k_cvtx(const float* __restrict__ x, u16* __restrict__ xbf_t) {
    __shared__ u16 tile[Dd][68];            // [e][m_local], padded stride
    const int b  = blockIdx.y;
    const int m0 = blockIdx.x * 64;
    const int t  = threadIdx.x;
    {   // read 64 rows x 128 cols, coalesced; scatter-transpose into LDS
        const int m = t >> 2, e0 = (t & 3) * 32;
        const float4* src = (const float4*)(x + ((size_t)b * Nn + m0 + m) * Dd + e0);
#pragma unroll
        for (int q = 0; q < 8; ++q) {
            float4 v = src[q];
            tile[e0 + q * 4 + 0][m] = f2bf(v.x);
            tile[e0 + q * 4 + 1][m] = f2bf(v.y);
            tile[e0 + q * 4 + 2][m] = f2bf(v.z);
            tile[e0 + q * 4 + 3][m] = f2bf(v.w);
        }
    }
    __syncthreads();
    {   // write contiguous 64B runs per thread
        const int e = t >> 1, h = t & 1;
        u16* dst = xbf_t + ((size_t)b * Dd + e) * Nn + m0 + h * 32;
#pragma unroll
        for (int i = 0; i < 32; ++i) dst[i] = tile[e][h * 32 + i];
    }
}

// ---------------------------------------------------------------------------
// 3) tiny coefficient heads
__global__ void k_coeffs(const float* __restrict__ ebar,
                         const float* __restrict__ Wa, const float* __restrict__ ba,
                         const float* __restrict__ Wb, const float* __restrict__ bb,
                         const float* __restrict__ Ws, const float* __restrict__ bs,
                         float* __restrict__ aC, float* __restrict__ bC,
                         float* __restrict__ sC) {
    const int t = threadIdx.x;
    if (t >= 30) return;
    const int b = t / 15, rem = t % 15, kind = rem / 5, r = rem % 5;
    const float* W  = (kind == 0) ? Wa : (kind == 1) ? Wb : Ws;
    const float* bi = (kind == 0) ? ba : (kind == 1) ? bb : bs;
    float acc = bi[r];
    for (int d = 0; d < Dd; ++d) acc += ebar[b * Dd + d] * W[d * 5 + r];
    if (kind == 0)      aC[b * RHO + r] = acc;
    else if (kind == 1) bC[b * RHO + r] = acc;
    else                sC[b * Jj + r] = 5.0f / (1.0f + __expf(-acc));
}

// ---------------------------------------------------------------------------
// 4) Wcat'[c][e] = sum_f P[c][f]*Ww[f][e], stored transposed bf16 [e][c];
//    block 768 computes folded bias cvec = (bS + sum_j bM_j) @ Ww + bw
__global__ void k_foldW(const float* __restrict__ WS, const float* __restrict__ WM,
                        const float* __restrict__ Ww, const float* __restrict__ bS,
                        const float* __restrict__ bM, const float* __restrict__ bw,
                        u16* __restrict__ WcatB, float* __restrict__ cvec) {
    const int e = threadIdx.x;
    const int c = blockIdx.x;
    if (c == Cc) {
        float acc = bw[e];
        for (int f = 0; f < Dd; ++f) {
            float cb = bS[f];
#pragma unroll
            for (int j = 0; j < Jj; ++j) cb += bM[j * Dd + f];
            acc += cb * Ww[f * Dd + e];
        }
        cvec[e] = acc;
        return;
    }
    const float* Prow;
    if (c < Dd) Prow = WS + (size_t)c * Dd;
    else {
        const int j = (c >> 7) - 1, d = c & (Dd - 1);
        Prow = WM + ((size_t)j * Dd + d) * Dd;
    }
    float acc = 0.0f;
    for (int f = 0; f < Dd; ++f) acc += Prow[f] * Ww[f * Dd + e];
    WcatB[(size_t)e * Cc + c] = f2bf(acc);
}

// ---------------------------------------------------------------------------
// WMMA fragment helpers (layouts per ISA 7.12.2: lanes 0-15 K{0..7,16..23},
// lanes 16-31 K{8..15,24..31})
__device__ __forceinline__ v16bf fragA(const u16* As, int row_base, int lane) {
    const int lrow = lane & 15;
    const int kofs = (lane & 16) ? 8 : 0;
    const u16* p = As + (row_base + lrow) * AS_STRIDE + kofs;
    FragU f;
    f.h[0] = *(const u16x8*)p;
    f.h[1] = *(const u16x8*)(p + 16);
    return f.v;
}
__device__ __forceinline__ v16bf fragG(const u16* p) {  // global, pre-offset
    FragU f;
    f.h[0] = *(const u16x8*)p;
    f.h[1] = *(const u16x8*)(p + 16);
    return f.v;
}

// ---------------------------------------------------------------------------
// 5) GEMM1: xt[b] = U[b]^T @ x[b]   (M=n, Ncol=d, K=m)
//    A (U^T) staged to LDS transposed w/ packed bf16 cvt; pipelined prefetch.
__global__ void k_gemm1(const float* __restrict__ U, const u16* __restrict__ xbf_t,
                        float* __restrict__ xt) {
    const int b = blockIdx.y, n0 = blockIdx.x * 128;
    const int tid = threadIdx.x, lane = tid & 31, wid = tid >> 5;
    const int wm = wid >> 1, wn = wid & 1;
    __shared__ __align__(16) u16 As[128 * AS_STRIDE];
    uint32_t* As32 = (uint32_t*)As;
    f32x8 acc[2][4];
#pragma unroll
    for (int a = 0; a < 2; ++a)
#pragma unroll
        for (int c = 0; c < 4; ++c) acc[a][c] = (f32x8){};

    const float* Ub = U + (size_t)b * Nn * Nn;
    const int lrow = lane & 15;
    const int kofs = (lane & 16) ? 8 : 0;
    const u16* Bp[4];
#pragma unroll
    for (int cs = 0; cs < 4; ++cs) {
        const int col = wn * 64 + cs * 16 + lrow;
        Bp[cs] = xbf_t + ((size_t)b * Dd + col) * Nn + kofs;
    }
    // stage mapping: each thread owns k columns {kk,kk+1} for 8 rows r0..r0+7
    const int kk = (tid >> 4) * 2;        // 0,2,...,30
    const int r0 = (tid & 15) * 8;        // 0..120
    float ra[8], rb[8];
    {   // prologue: prefetch slab k0=0
        const float* s0 = Ub + (size_t)kk * Nn + n0 + r0;
#pragma unroll
        for (int i = 0; i < 8; ++i) { ra[i] = s0[i]; rb[i] = s0[Nn + i]; }
    }
    for (int k0 = 0; k0 < Nn; k0 += 32) {
        __syncthreads();                  // prior compute done reading LDS
        {   // packed transpose store: one b32 per row
#pragma unroll
            for (int i = 0; i < 8; ++i)
                As32[(r0 + i) * (AS_STRIDE / 2) + (kk >> 1)] = pack_bf2(ra[i], rb[i]);
        }
        if (k0 + 32 < Nn) {               // prefetch next slab during WMMA
            const float* s0 = Ub + (size_t)(k0 + 32 + kk) * Nn + n0 + r0;
#pragma unroll
            for (int i = 0; i < 8; ++i) { ra[i] = s0[i]; rb[i] = s0[Nn + i]; }
        }
        __syncthreads();
        v16bf a0 = fragA(As, wm * 32, lane);
        v16bf a1 = fragA(As, wm * 32 + 16, lane);
#pragma unroll
        for (int cs = 0; cs < 4; ++cs) {
            v16bf bfr = fragG(Bp[cs] + k0);
            acc[0][cs] = wmma_bf16(a0, bfr, acc[0][cs]);
            acc[1][cs] = wmma_bf16(a1, bfr, acc[1][cs]);
        }
    }
    const int rbase = n0 + wm * 32 + ((lane & 16) ? 8 : 0);
#pragma unroll
    for (int rs = 0; rs < 2; ++rs)
#pragma unroll
        for (int cs = 0; cs < 4; ++cs) {
            const int col = wn * 64 + cs * 16 + lrow;
#pragma unroll
            for (int v = 0; v < 8; ++v)
                xt[((size_t)b * Nn + rbase + rs * 16 + v) * Dd + col] = acc[rs][cs][v];
        }
}

// ---------------------------------------------------------------------------
// 6) z = Fcat @ Wcat' with Fcat built on the fly (Chebyshev filters * xt).
//    Output stored transposed bf16: zbf_t[(b*128+e)*4096 + n].
__global__ void k_filter_zgemm(const float* __restrict__ eigvs, const float* __restrict__ xt,
                               const float* __restrict__ aC, const float* __restrict__ bC,
                               const float* __restrict__ sC, const u16* __restrict__ WcatB,
                               u16* __restrict__ zbf_t) {
    const int b = blockIdx.y, n0 = blockIdx.x * 128;
    const int tid = threadIdx.x, lane = tid & 31, wid = tid >> 5;
    const int wm = wid >> 1, wn = wid & 1;
    __shared__ __align__(16) u16 Fs[128 * AS_STRIDE];
    uint32_t* Fs32 = (uint32_t*)Fs;
    float ac[RHO], bc[RHO], sc[Jj];
#pragma unroll
    for (int i = 0; i < RHO; ++i) { ac[i] = aC[b * RHO + i]; bc[i] = bC[b * RHO + i]; }
#pragma unroll
    for (int i = 0; i < Jj; ++i) sc[i] = sC[b * Jj + i];

    f32x8 acc[2][4];
#pragma unroll
    for (int a = 0; a < 2; ++a)
#pragma unroll
        for (int c = 0; c < 4; ++c) acc[a][c] = (f32x8){};

    const int lrow = lane & 15;
    const int kofs = (lane & 16) ? 8 : 0;
    const u16* Bp[4];
#pragma unroll
    for (int cs = 0; cs < 4; ++cs) {
        const int col = wn * 64 + cs * 16 + lrow;
        Bp[cs] = WcatB + (size_t)col * Cc + kofs;
    }
    const int sr = tid >> 1, sh = tid & 1;
    float ev[16], xv[16];
    {   // prologue prefetch (c0 = 0 -> d0 = sh*16)
        const size_t ro = ((size_t)b * Nn + n0 + sr) * Dd + sh * 16;
#pragma unroll
        for (int i = 0; i < 16; ++i) { ev[i] = eigvs[ro + i]; xv[i] = xt[ro + i]; }
    }
    for (int c0 = 0; c0 < Cc; c0 += 32) {
        const int j = c0 >> 7;            // 0 => h filter, 1..5 => g_{j-1}
        const float sj = (j == 0) ? 0.0f : sc[j - 1];
        __syncthreads();
        {   // filter + pack + 2x b128 LDS store
            uint32_t pk[8];
#pragma unroll
            for (int i = 0; i < 8; ++i) {
                const float f0 = (j == 0) ? cheb_h(ev[2 * i], bc)
                                          : cheb_g(sj * ev[2 * i], ac);
                const float f1 = (j == 0) ? cheb_h(ev[2 * i + 1], bc)
                                          : cheb_g(sj * ev[2 * i + 1], ac);
                pk[i] = pack_bf2(f0 * xv[2 * i], f1 * xv[2 * i + 1]);
            }
            uint4* dst = (uint4*)&Fs32[sr * (AS_STRIDE / 2) + sh * 8];
            dst[0] = make_uint4(pk[0], pk[1], pk[2], pk[3]);
            dst[1] = make_uint4(pk[4], pk[5], pk[6], pk[7]);
        }
        if (c0 + 32 < Cc) {               // prefetch next slice's inputs
            const int d0 = ((c0 + 32) & (Dd - 1)) + sh * 16;
            const size_t ro = ((size_t)b * Nn + n0 + sr) * Dd + d0;
#pragma unroll
            for (int i = 0; i < 16; ++i) { ev[i] = eigvs[ro + i]; xv[i] = xt[ro + i]; }
        }
        __syncthreads();
        v16bf a0 = fragA(Fs, wm * 32, lane);
        v16bf a1 = fragA(Fs, wm * 32 + 16, lane);
#pragma unroll
        for (int cs = 0; cs < 4; ++cs) {
            v16bf bfr = fragG(Bp[cs] + c0);
            acc[0][cs] = wmma_bf16(a0, bfr, acc[0][cs]);
            acc[1][cs] = wmma_bf16(a1, bfr, acc[1][cs]);
        }
    }
    const int rbase = n0 + wm * 32 + ((lane & 16) ? 8 : 0);
#pragma unroll
    for (int rs = 0; rs < 2; ++rs)
#pragma unroll
        for (int cs = 0; cs < 4; ++cs) {
            const int col = wn * 64 + cs * 16 + lrow;
#pragma unroll
            for (int v = 0; v < 8; ++v)
                zbf_t[((size_t)b * Dd + col) * Nn + rbase + rs * 16 + v] =
                    f2bf(acc[rs][cs][v]);
        }
}

// ---------------------------------------------------------------------------
// 7) GEMM2: out[b] = U[b] @ z[b] + cvec   (A = U row-major, pipelined staging)
__global__ void k_gemm2(const float* __restrict__ U, const u16* __restrict__ zbf_t,
                        const float* __restrict__ cvec, float* __restrict__ out) {
    const int b = blockIdx.y, n0 = blockIdx.x * 128;
    const int tid = threadIdx.x, lane = tid & 31, wid = tid >> 5;
    const int wm = wid >> 1, wn = wid & 1;
    __shared__ __align__(16) u16 As[128 * AS_STRIDE];
    uint32_t* As32 = (uint32_t*)As;
    f32x8 acc[2][4];
#pragma unroll
    for (int a = 0; a < 2; ++a)
#pragma unroll
        for (int c = 0; c < 4; ++c) acc[a][c] = (f32x8){};

    const float* Ub = U + (size_t)b * Nn * Nn;
    const int lrow = lane & 15;
    const int kofs = (lane & 16) ? 8 : 0;
    const u16* Bp[4];
#pragma unroll
    for (int cs = 0; cs < 4; ++cs) {
        const int col = wn * 64 + cs * 16 + lrow;
        Bp[cs] = zbf_t + ((size_t)b * Dd + col) * Nn + kofs;
    }
    const int sr = tid >> 1, sh = tid & 1;   // row, 16-col half
    float4 q[4];
    {   // prologue prefetch
        const float4* s = (const float4*)(Ub + (size_t)(n0 + sr) * Nn + sh * 16);
#pragma unroll
        for (int i = 0; i < 4; ++i) q[i] = s[i];
    }
    for (int k0 = 0; k0 < Nn; k0 += 32) {
        __syncthreads();
        {   // packed contiguous store: 2x b128
            uint32_t pk[8];
#pragma unroll
            for (int i = 0; i < 4; ++i) {
                pk[2 * i]     = pack_bf2(q[i].x, q[i].y);
                pk[2 * i + 1] = pack_bf2(q[i].z, q[i].w);
            }
            uint4* dst = (uint4*)&As32[sr * (AS_STRIDE / 2) + sh * 8];
            dst[0] = make_uint4(pk[0], pk[1], pk[2], pk[3]);
            dst[1] = make_uint4(pk[4], pk[5], pk[6], pk[7]);
        }
        if (k0 + 32 < Nn) {               // prefetch next slab during WMMA
            const float4* s =
                (const float4*)(Ub + (size_t)(n0 + sr) * Nn + k0 + 32 + sh * 16);
#pragma unroll
            for (int i = 0; i < 4; ++i) q[i] = s[i];
        }
        __syncthreads();
        v16bf a0 = fragA(As, wm * 32, lane);
        v16bf a1 = fragA(As, wm * 32 + 16, lane);
#pragma unroll
        for (int cs = 0; cs < 4; ++cs) {
            v16bf bfr = fragG(Bp[cs] + k0);
            acc[0][cs] = wmma_bf16(a0, bfr, acc[0][cs]);
            acc[1][cs] = wmma_bf16(a1, bfr, acc[1][cs]);
        }
    }
    const int rbase = n0 + wm * 32 + ((lane & 16) ? 8 : 0);
#pragma unroll
    for (int rs = 0; rs < 2; ++rs)
#pragma unroll
        for (int cs = 0; cs < 4; ++cs) {
            const int col = wn * 64 + cs * 16 + lrow;
            const float cadd = cvec[col];
#pragma unroll
            for (int v = 0; v < 8; ++v)
                out[((size_t)b * Nn + rbase + rs * 16 + v) * Dd + col] =
                    acc[rs][cs][v] + cadd;
        }
}

// ---------------------------------------------------------------------------
// Workspace layout (bytes)
#define EBAR_OFF   ((size_t)0)          // 2*128 f32
#define ACOEF_OFF  ((size_t)1024)       // 2*5 f32
#define BCOEF_OFF  ((size_t)1088)
#define SCOEF_OFF  ((size_t)1152)
#define CVEC_OFF   ((size_t)1280)       // 128 f32
#define WCAT_OFF   ((size_t)4096)       // 128*768 bf16 (transposed [e][c])
#define XBF_OFF    ((size_t)200704)     // 2*128*4096 bf16 (transposed [b][e][m])
#define XT_OFF     ((size_t)2297856)    // 2*4096*128 f32
#define ZBF_OFF    ((size_t)6492160)    // 2*128*4096 bf16 (transposed)
// total ~8.6 MB

extern "C" void kernel_launch(void* const* d_in, const int* in_sizes, int n_in,
                              void* d_out, int out_size, void* d_ws, size_t ws_size,
                              hipStream_t stream) {
    (void)in_sizes; (void)n_in; (void)out_size; (void)ws_size;
    const float* eigvs = (const float*)d_in[0];
    const float* x     = (const float*)d_in[1];
    const float* U     = (const float*)d_in[2];
    const float* Wa    = (const float*)d_in[3];
    const float* ba    = (const float*)d_in[4];
    const float* Wb    = (const float*)d_in[5];
    const float* bb    = (const float*)d_in[6];
    const float* Ws    = (const float*)d_in[7];
    const float* bs    = (const float*)d_in[8];
    const float* WS    = (const float*)d_in[9];
    const float* bS    = (const float*)d_in[10];
    const float* WM    = (const float*)d_in[11];
    const float* bM    = (const float*)d_in[12];
    const float* Ww    = (const float*)d_in[13];
    const float* bw    = (const float*)d_in[14];

    char* ws = (char*)d_ws;
    float* ebar  = (float*)(ws + EBAR_OFF);
    float* aC    = (float*)(ws + ACOEF_OFF);
    float* bC    = (float*)(ws + BCOEF_OFF);
    float* sC    = (float*)(ws + SCOEF_OFF);
    float* cvec  = (float*)(ws + CVEC_OFF);
    u16*   WcatB = (u16*)(ws + WCAT_OFF);
    u16*   xbf_t = (u16*)(ws + XBF_OFF);
    float* xt    = (float*)(ws + XT_OFF);
    u16*   zbf_t = (u16*)(ws + ZBF_OFF);
    float* out   = (float*)d_out;

    k_colmean<<<dim3(Bsz), dim3(Dd), 0, stream>>>(eigvs, ebar);
    k_cvtx<<<dim3(Nn / 64, Bsz), dim3(256), 0, stream>>>(x, xbf_t);
    k_coeffs<<<dim3(1), dim3(32), 0, stream>>>(ebar, Wa, ba, Wb, bb, Ws, bs, aC, bC, sC);
    k_foldW<<<dim3(Cc + 1), dim3(Dd), 0, stream>>>(WS, WM, Ww, bS, bM, bw, WcatB, cvec);
    k_gemm1<<<dim3(Nn / 128, Bsz), dim3(256), 0, stream>>>(U, xbf_t, xt);
    k_filter_zgemm<<<dim3(Nn / 128, Bsz), dim3(256), 0, stream>>>(eigvs, xt, aC, bC, sC,
                                                                  WcatB, zbf_t);
    k_gemm2<<<dim3(Nn / 128, Bsz), dim3(256), 0, stream>>>(U, zbf_t, cvec, out);
}
